// BitwisePopcountLinear_6743098655625
// MI455X (gfx1250) — 8-bit matrix operand order repaired and checked by reading, compiled, not run on hardware
//
#include <hip/hip_runtime.h>
#include <hip/hip_bf16.h>


typedef __attribute__((ext_vector_type(8))) int v8i;

#define THR 0.05f

__device__ __forceinline__ int dot4u8(unsigned a, unsigned b, int c)
{
#if __has_builtin(__builtin_amdgcn_udot4)
    return (int)__builtin_amdgcn_udot4(a, b, (unsigned)c, false);
#else
    int s = c;
    s += (int)((a       ) & 0xffu) * (int)((b       ) & 0xffu);
    s += (int)((a >>  8u) & 0xffu) * (int)((b >>  8u) & 0xffu);
    s += (int)((a >> 16u) & 0xffu) * (int)((b >> 16u) & 0xffu);
    s += (int)( a >> 24u         ) * (int)( b >> 24u         );
    return s;
#endif
}

__device__ __forceinline__ void pack_weighted(float4 v, unsigned& lo, unsigned& hi)
{
    unsigned n0 = v.x <= -THR, p0 = v.x >= THR;
    unsigned n1 = v.y <= -THR, p1 = v.y >= THR;
    unsigned n2 = v.z <= -THR, p2 = v.z >= THR;
    unsigned n3 = v.w <= -THR, p3 = v.w >= THR;
    lo = (n0 * 128u) | ((p0 * 64u) << 8) | ((n1 * 32u) << 16) | ((p1 * 16u) << 24);
    hi = (n2 *   8u) | ((p2 *  4u) << 8) | ((n3 *  2u) << 16) | ((p3 *  1u) << 24);
}

__device__ __forceinline__ void pack_bits(float4 v, unsigned& lo, unsigned& hi)
{
    unsigned n0 = v.x <= -THR, p0 = v.x >= THR;
    unsigned n1 = v.y <= -THR, p1 = v.y >= THR;
    unsigned n2 = v.z <= -THR, p2 = v.z >= THR;
    unsigned n3 = v.w <= -THR, p3 = v.w >= THR;
    lo = n0 | (p0 << 8) | (n1 << 16) | (p1 << 24);
    hi = n2 | (p2 << 8) | (n3 << 16) | (p3 << 24);
}

__global__ __launch_bounds__(256)
void quant_x_kernel(const float* __restrict__ x, unsigned char* __restrict__ a8,
                    float* __restrict__ sx)
{
    __shared__ int red[256];
    const int row = blockIdx.x;
    const int t   = threadIdx.x;
    const float*   src = x  + (size_t)row * 4096 + t * 16;
    unsigned char* dst = a8 + (size_t)row * 8192 + t * 32;

    int acc = 0;
#pragma unroll
    for (int i = 0; i < 4; ++i) {
        float4 v = *(const float4*)(src + i * 4);
        unsigned lo, hi;
        pack_weighted(v, lo, hi);
        ((uint2*)dst)[i] = make_uint2(lo, hi);
        acc = dot4u8(lo, 0x01010101u, acc);
        acc = dot4u8(hi, 0x01010101u, acc);
    }
    red[t] = acc;
    __syncthreads();
    for (int off = 128; off > 0; off >>= 1) {
        if (t < off) red[t] += red[t + off];
        __syncthreads();
    }
    if (t == 0) sx[row] = (float)red[0];
}

__global__ __launch_bounds__(256)
void ternary_gemm_wmma(const unsigned char* __restrict__ a8,
                       const float* __restrict__ w,
                       const float* __restrict__ sx,
                       float* __restrict__ out)
{
    constexpr int F    = 4096;
    constexpr int C    = 4096;
    constexpr int BN   = 32;
    constexpr int FC   = 32;
    constexpr int NK   = F / FC;
    constexpr int BSTR = 72;

    __shared__ unsigned char ldsB[2 * BN * BSTR];
    __shared__ int swRed[BN * 8];

    const int t      = threadIdx.x;
    const int bn     = blockIdx.x * BN;
    const int wave   = t >> 5;
    const int lane   = t & 31;
    const int laneLo = lane & 15;
    const int laneHi = lane >> 4;
    const int wm     = (wave >> 1) * 64;
    const int wn     = (wave & 1) * 16;

    const int wrow = t >> 3;
    const int wf4  = t & 7;

    v8i acc[4] = {};
    int swAcc = 0;
    float4 wr;

    auto wload = [&](int kc) {
        const int fb = kc * FC;
        wr = *(const float4*)(w + (size_t)(bn + wrow) * F + fb + wf4 * 4);
    };

    auto qstoreB = [&](unsigned char* B) {
        unsigned lo, hi;
        pack_bits(wr, lo, hi);
        unsigned* dst = (unsigned*)(B + wrow * BSTR + wf4 * 8);
        dst[0] = lo;
        dst[1] = hi;
        swAcc = dot4u8(lo, 0x10204080u, swAcc);
        swAcc = dot4u8(hi, 0x01020408u, swAcc);
    };

    auto afragG = [&](int kc, int m0) -> v8i {
        v8i r;
        const unsigned char* base =
            a8 + (size_t)(m0 + laneLo) * 8192 + kc * 64 + laneHi * 8;
#pragma unroll
        for (int p = 0; p < 4; ++p) {
            uint2 d = *(const uint2*)(base + p * 16);
            r[2 * p]     = (int)d.x;
            r[2 * p + 1] = (int)d.y;
        }
        return r;
    };

    auto bfrag = [&](const unsigned char* B, int n0) -> v8i {
        v8i r;
        const unsigned char* base = B + (n0 + laneLo) * BSTR + laneHi * 8;
#pragma unroll
        for (int j = 0; j < 8; ++j) {
            int k = ((j >> 1) << 4) | ((j & 1) << 2);
            r[j] = *(const int*)(base + k);
        }
        return r;
    };

    wload(0);
    for (int kc = 0; kc < NK; ++kc) {
        unsigned char* B = ldsB + (kc & 1) * (BN * BSTR);
        qstoreB(B);
        __syncthreads();
        if (kc + 1 < NK) wload(kc + 1);

        v8i bv = bfrag(B, wn);
#pragma unroll
        for (int sm = 0; sm < 4; ++sm) {
            v8i av = afragG(kc, wm + sm * 16);
            acc[sm] = __builtin_amdgcn_wmma_i32_16x16x64_iu8(
                false, av, false, bv, acc[sm], false, false);
        }
    }

    swRed[wrow * 8 + wf4] = swAcc;
    __syncthreads();

    const int colLocal = wn + laneLo;
    int swi = 0;
#pragma unroll
    for (int j = 0; j < 8; ++j) swi += swRed[colLocal * 8 + j];
    const float swv = (float)swi;
    const int   col = bn + colLocal;

#pragma unroll
    for (int sm = 0; sm < 4; ++sm) {
#pragma unroll
        for (int r = 0; r < 8; ++r) {
            int row = wm + sm * 16 + r + 8 * laneHi;
            float cross = (float)acc[sm][r];
            out[(size_t)row * C + col] = 8192.0f - sx[row] - swv + 2.0f * cross;
        }
    }
}

extern "C" void kernel_launch(void* const* d_in, const int* in_sizes, int n_in,
                              void* d_out, int out_size, void* d_ws, size_t ws_size,
                              hipStream_t stream)
{
    (void)in_sizes; (void)n_in; (void)out_size; (void)ws_size;
    const float* x = (const float*)d_in[0];
    const float* w = (const float*)d_in[1];
    float* out = (float*)d_out;

    float*         sx = (float*)d_ws;
    unsigned char* a8 = (unsigned char*)d_ws + 1024;

    quant_x_kernel<<<256, 256, 0, stream>>>(x, a8, sx);
    ternary_gemm_wmma<<<4096 / 32, 256, 0, stream>>>(a8, w, sx, out);
}
